// ArcFaceHead_13297218748568
// MI455X (gfx1250) — matrix operand order repaired and checked by reading, not compiled, not run on hardware
//
#include <hip/hip_runtime.h>
#include <hip/hip_bf16.h>
#include <math.h>

// ---------------------------------------------------------------------------
// ArcFace head for MI455X (gfx1250, wave32, WMMA + async global->LDS).
//   emb (1024x512 f32), labels (1024 i32), weight (100000x512 f32)
//   out (1024x100000 f32)
// Pipeline: row-L2-normalize A and W into f16 (workspace), then a register-
// blocked f16 WMMA GEMM (wave tile 32x64 = 2x4 WMMA tiles, block 64x256),
// K staged through double-buffered LDS via GLOBAL_LOAD_ASYNC_TO_LDS_B128
// (no VGPR staging -> no spills), fused ArcFace margin epilogue with
// non-temporal logit stores (keeps f16 weights resident in the 192MB L2).
// ---------------------------------------------------------------------------

#define EMBED_DIM   512
#define NUM_CLASSES 100000
#define BATCH       1024

#define ARC_SCALE 64.0f
#define ARC_COS_M 0.8775825618903728f   // cos(0.5)
#define ARC_SIN_M 0.47942553860420301f  // sin(0.5)
#define ARC_TH   (-0.8775825618903728f) // cos(pi-0.5)
#define ARC_MM    0.23971276930210150f  // sin(pi-0.5)*0.5

typedef _Float16 v16h __attribute__((ext_vector_type(16)));
typedef _Float16 v8h  __attribute__((ext_vector_type(8)));
typedef _Float16 v4h  __attribute__((ext_vector_type(4)));
typedef float    v8f  __attribute__((ext_vector_type(8)));

// Async 16B global->LDS copy (per lane). VDST = LDS byte address (low 32 bits
// of the flat shared-aperture address), VADDR = 64-bit global address.
// Tracked by ASYNCcnt; completes in order with other async loads.
__device__ __forceinline__ void async_cp16(void* ldsPtr, const void* gPtr) {
  const unsigned ldsOff = (unsigned)(uintptr_t)ldsPtr;
  asm volatile("global_load_async_to_lds_b128 %0, %1, off"
               :: "v"(ldsOff), "v"(gPtr)
               : "memory");
}
__device__ __forceinline__ void wait_async0() {
  asm volatile("s_wait_asynccnt 0" ::: "memory");
}

// ---------------------------------------------------------------------------
// Kernels 1/2: L2-normalize rows of a (nrows x 512) f32 matrix into f16.
// One wave32 per row: 16 elements/lane (4x float4 in 512B wave bursts),
// square-sum, 5-step shfl_xor reduction, scale + convert.
// ---------------------------------------------------------------------------
__global__ __launch_bounds__(256) void rownorm512_f32_to_f16(
    const float* __restrict__ src, _Float16* __restrict__ dst, int nrows) {
  const int wavesPerBlk = blockDim.x >> 5;
  const int row  = blockIdx.x * wavesPerBlk + (threadIdx.x >> 5);
  const int lane = threadIdx.x & 31;
  if (row >= nrows) return;

  const float4* __restrict__ r4 =
      (const float4*)(src + (size_t)row * EMBED_DIM);
  float4 v[4];
  float n2 = 0.0f;
#pragma unroll
  for (int i = 0; i < 4; ++i) {
    v[i] = r4[lane + i * 32];
    n2 = fmaf(v[i].x, v[i].x, n2);
    n2 = fmaf(v[i].y, v[i].y, n2);
    n2 = fmaf(v[i].z, v[i].z, n2);
    n2 = fmaf(v[i].w, v[i].w, n2);
  }
#pragma unroll
  for (int off = 16; off >= 1; off >>= 1) n2 += __shfl_xor(n2, off, 32);

  const float inv = 1.0f / fmaxf(sqrtf(n2), 1e-12f);

  v4h* __restrict__ d4 = (v4h*)(dst + (size_t)row * EMBED_DIM);
#pragma unroll
  for (int i = 0; i < 4; ++i) {
    v4h h;
    h.x = (_Float16)(v[i].x * inv);
    h.y = (_Float16)(v[i].y * inv);
    h.z = (_Float16)(v[i].z * inv);
    h.w = (_Float16)(v[i].w * inv);
    d4[lane + i * 32] = h;
  }
}

// ---------------------------------------------------------------------------
// Kernel 3: register-blocked WMMA GEMM + ArcFace epilogue.
//   Block: 256 threads = 8 waves, arranged 2(M) x 4(N).
//   Block tile: 64(M) x 256(N). Wave tile: 32x64 = 2x4 WMMA 16x16 tiles.
//   K staged in chunks of 64 via async global->LDS, double-buffered,
//   one barrier + one s_wait_asynccnt per stage.
//   Per K=32 substep: 2 A-frags + 4 B-frags -> 8 v_wmma_f32_16x16x32_f16.
// ---------------------------------------------------------------------------
#define MBLK 64
#define NBLK 256
#define KBLK 64
#define LDSP 72   // padded row length in halves (+16B) to spread LDS banks

__global__ __launch_bounds__(256) void arcface_wmma_gemm(
    const _Float16* __restrict__ A,   // 1024 x 512, L2-normalized, f16
    const _Float16* __restrict__ W,   // 100000 x 512, L2-normalized, f16
    const int* __restrict__ labels,   // 1024
    float* __restrict__ out) {        // 1024 x 100000
  __shared__ _Float16 sA[2][MBLK][LDSP];   // 18 KB
  __shared__ _Float16 sB[2][NBLK][LDSP];   // 72 KB
  __shared__ int sLab[MBLK];

  const int t     = threadIdx.x;
  const int nBase = blockIdx.x * NBLK;
  const int mBase = blockIdx.y * MBLK;

  if (t < MBLK) sLab[t] = labels[mBase + t];

  const int wave = t >> 5;
  const int lane = t & 31;
  const int lr   = lane & 15;
  const bool hi  = lane >= 16;
  const int wmI  = wave >> 2;           // 0..1 -> M offset wmI*32
  const int wnI  = wave & 3;            // 0..3 -> N offset wnI*64

  // Async staging: each thread moves 2x16B of A and 4 rows x 2x16B of B
  // per K=64 stage, global -> LDS with no VGPR staging.
  const int a_r = t >> 2;               // 0..63
  const int a_c = (t & 3) * 16;         // halves: 0,16,32,48
  const _Float16* __restrict__ aRow = A + (size_t)(mBase + a_r) * EMBED_DIM;
  const _Float16* __restrict__ bRow[4];
#pragma unroll
  for (int i = 0; i < 4; ++i) {
    int wr = nBase + a_r + 64 * i;
    if (wr >= NUM_CLASSES) wr = NUM_CLASSES - 1;   // clamp; stores guarded
    bRow[i] = W + (size_t)wr * EMBED_DIM;
  }

  auto issueStage = [&](int ko, int p) {
    async_cp16(&sA[p][a_r][a_c],     aRow + ko + a_c);
    async_cp16(&sA[p][a_r][a_c + 8], aRow + ko + a_c + 8);
#pragma unroll
    for (int i = 0; i < 4; ++i) {
      async_cp16(&sB[p][a_r + 64 * i][a_c],     bRow[i] + ko + a_c);
      async_cp16(&sB[p][a_r + 64 * i][a_c + 8], bRow[i] + ko + a_c + 8);
    }
  };

  v8f acc[2][4];
#pragma unroll
  for (int mi = 0; mi < 2; ++mi)
#pragma unroll
    for (int ni = 0; ni < 4; ++ni)
      acc[mi][ni] = (v8f){0.f, 0.f, 0.f, 0.f, 0.f, 0.f, 0.f, 0.f};

  constexpr int NSTAGE = EMBED_DIM / KBLK;   // 8

  issueStage(0, 0);

  for (int k = 0; k < NSTAGE; ++k) {
    const int p = k & 1;
    wait_async0();       // our stage-k (and any earlier) transfers landed
    __syncthreads();     // all waves' stage-k data visible; all waves are
                         // done computing stage k-1 (same buffer as k+1)
    if (k + 1 < NSTAGE)  // overlap: stage k+1 streams in during compute k
      issueStage((k + 1) * KBLK, p ^ 1);

#pragma unroll
    for (int s = 0; s < 2; ++s) {    // two K=32 substeps
      const int kb = s * 32;
      // A fragment (16x32 f16): lanes 0-15 hold K{0..7,16..23} of row M=lr,
      // lanes 16-31 hold K{8..15,24..31}.
      v16h af[2];
#pragma unroll
      for (int mi = 0; mi < 2; ++mi) {
        const int ar = wmI * 32 + mi * 16 + lr;
        const v8h lo = *(const v8h*)&sA[p][ar][kb + (hi ? 8 : 0)];
        const v8h hv = *(const v8h*)&sA[p][ar][kb + (hi ? 24 : 16)];
        af[mi] = __builtin_shufflevector(
            lo, hv, 0, 1, 2, 3, 4, 5, 6, 7, 8, 9, 10, 11, 12, 13, 14, 15);
      }
      // B fragment (32x16 f16): lanes 0-15 hold K{0..7,16..23} of col N=lr,
      // lanes 16-31 hold K{8..15,24..31}.
      v16h bf[4];
#pragma unroll
      for (int ni = 0; ni < 4; ++ni) {
        const int br = wnI * 64 + ni * 16 + lr;
        const v8h lo = *(const v8h*)&sB[p][br][kb + (hi ? 8 : 0)];
        const v8h hv = *(const v8h*)&sB[p][br][kb + (hi ? 24 : 16)];
        bf[ni] = __builtin_shufflevector(
            lo, hv, 0, 1, 2, 3, 4, 5, 6, 7, 8, 9, 10, 11, 12, 13, 14, 15);
      }
#pragma unroll
      for (int mi = 0; mi < 2; ++mi)
#pragma unroll
        for (int ni = 0; ni < 4; ++ni)
          acc[mi][ni] = __builtin_amdgcn_wmma_f32_16x16x32_f16(
              false, af[mi], false, bf[ni], (short)0, acc[mi][ni],
              false, false);
    }
  }

  // Epilogue. C/D layout: lanes 0-15: VGPR r = (M=r, N=lane);
  //                       lanes 16-31: VGPR r = (M=8+r, N=lane-16).
  // Non-temporal stores: the 410MB logit stream must not evict the
  // L2-resident f16 weights.
#pragma unroll
  for (int ni = 0; ni < 4; ++ni) {
    const int col = nBase + wnI * 64 + ni * 16 + lr;
    if (col >= NUM_CLASSES) continue;
#pragma unroll
    for (int mi = 0; mi < 2; ++mi) {
      const int mOff = wmI * 32 + mi * 16 + (hi ? 8 : 0);
#pragma unroll
      for (int r = 0; r < 8; ++r) {
        float c = acc[mi][ni][r];
        c = fminf(fmaxf(c, -1.0f + 1e-7f), 1.0f - 1e-7f);
        const float s   = sqrtf(1.0f - c * c);
        const float cm  = fmaf(c, ARC_COS_M, -(s * ARC_SIN_M));
        const float phi = (c > ARC_TH) ? cm : (c - ARC_MM);
        const int   lab = sLab[mOff + r];
        const float o   = ((lab == col) ? phi : c) * ARC_SCALE;
        __builtin_nontemporal_store(
            o, &out[(size_t)(mBase + mOff + r) * NUM_CLASSES + col]);
      }
    }
  }
}

// ---------------------------------------------------------------------------
// Launch
// ---------------------------------------------------------------------------
extern "C" void kernel_launch(void* const* d_in, const int* in_sizes, int n_in,
                              void* d_out, int out_size, void* d_ws,
                              size_t ws_size, hipStream_t stream) {
  const float* emb    = (const float*)d_in[0];  // (1024, 512)
  const int*   labels = (const int*)d_in[1];    // (1024,)
  const float* weight = (const float*)d_in[2];  // (100000, 512)
  float*       out    = (float*)d_out;          // (1024, 100000)

  // Workspace: [ f16 normalized A : 1 MB ][ f16 normalized W : 102.4 MB ]
  _Float16* embH = (_Float16*)d_ws;
  _Float16* wH   = (_Float16*)((char*)d_ws +
                               (size_t)BATCH * EMBED_DIM * sizeof(_Float16));

  rownorm512_f32_to_f16<<<(BATCH + 7) / 8, 256, 0, stream>>>(emb, embH, BATCH);
  rownorm512_f32_to_f16<<<(NUM_CLASSES + 7) / 8, 256, 0, stream>>>(
      weight, wH, NUM_CLASSES);

  dim3 grid((NUM_CLASSES + NBLK - 1) / NBLK, BATCH / MBLK);  // 391 x 16
  arcface_wmma_gemm<<<grid, 256, 0, stream>>>(embH, wH, labels, out);
}
